// MatchLayer_8667244003731
// MI455X (gfx1250) — matrix operand order repaired and checked by reading, not compiled, not run on hardware
//
#include <hip/hip_runtime.h>

// ---------------- problem dims ----------------
#define HH 256   // hidden
#define BB 64    // batch
#define QQ 64    // question length
#define PP 512   // passage length
// ---------------- launch config ----------------
#define NB 8               // cooperating blocks per direction
#define NT 512             // threads per block (16 waves, wave32)
#define NWAVE (NT / 32)
#define GW (NB * NWAVE)    // waves per direction = 128

typedef __attribute__((ext_vector_type(16))) __bf16 v16bf;
typedef __attribute__((ext_vector_type(8)))  float  v8f;

// ---- WMMA fragment helpers (contiguous k order k = 16*(lane/16) + s, same in A and B, wave32) ----
// A (16x32 bf16): lane m = lane&15; lanes<16 hold K 0-15; lanes>=16 hold K 16-31.
__device__ inline v16bf frag_a_f32(const float* row, int khalf) {
  v16bf a;
  const float* p = row + 2 * khalf;
#pragma unroll
  for (int i = 0; i < 8; ++i) {
    a[i]     = (__bf16)p[i];
    a[i + 8] = (__bf16)p[8 + i];
  }
  return a;
}
// B (32x16 bf16): lane n = lane&15; lanes<16 hold K 0-15, lanes>=16 hold K 16-31.
// Caller passes pointer already offset by col*K + kt*32 + kh16 (32B aligned).
__device__ inline v16bf frag_b_bf(const __bf16* p) {
  return *(const v16bf*)p;
}
__device__ inline v8f wmma_bf16(v16bf a, v16bf b, v8f c) {
  // (neg_a, A, neg_b, B, c_mod, C, reuse_a, reuse_b) -> v_wmma_f32_16x16x32_bf16
  return __builtin_amdgcn_wmma_f32_16x16x32_bf16(false, a, false, b, (short)0, c,
                                                 false, false);
}

__device__ inline float fast_exp(float x) { return __expf(x); }
__device__ inline float fast_sigmoid(float x) { return 1.f / (1.f + fast_exp(-x)); }
__device__ inline float fast_tanh(float x) {
  float e = fast_exp(2.f * x);          // inf-safe: e=inf -> 1, e=0 -> -1
  return 1.f - 2.f / (e + 1.f);
}

// ---- software grid barrier per direction (generation counter) ----
__device__ inline void dir_barrier(unsigned* s, int nb) {
  __syncthreads();
  if (threadIdx.x == 0) {
    __threadfence();
    volatile unsigned* gen = s + 1;
    unsigned g = *gen;
    unsigned prev = atomicAdd(s, 1u);
    if (prev == (unsigned)(nb - 1)) {
      atomicExch(s, 0u);
      __threadfence();
      atomicAdd((unsigned*)(s + 1), 1u);
    } else {
      while (*gen == g) { __builtin_amdgcn_s_sleep(1); }
    }
    __threadfence();
  }
  __syncthreads();
}

// ================= kernel 1: weight convert + state init =================
__global__ void k_prep(const float* whq, const float* whp, const float* whr,
                       const float* fwWih, const float* fwWhh,
                       const float* bwWih, const float* bwWhh,
                       __bf16* whq_bf, __bf16* whp_bf, __bf16* whr_bf,
                       __bf16* Wih_bf, __bf16* Whh_bf,
                       float* hx, float* cx, unsigned* sync) {
  int gt = blockIdx.x * blockDim.x + threadIdx.x;
  int stride = gridDim.x * blockDim.x;
  for (int i = gt; i < HH * HH; i += stride) {
    whq_bf[i] = (__bf16)whq[i];
    whp_bf[i] = (__bf16)whp[i];
    whr_bf[i] = (__bf16)whr[i];
  }
  for (int i = gt; i < 1024 * 512; i += stride) {
    Wih_bf[i]          = (__bf16)fwWih[i];
    Wih_bf[524288 + i] = (__bf16)bwWih[i];
  }
  for (int i = gt; i < 1024 * 256; i += stride) {
    Whh_bf[i]          = (__bf16)fwWhh[i];
    Whh_bf[262144 + i] = (__bf16)bwWhh[i];
  }
  for (int i = gt; i < 2 * BB * HH; i += stride) { hx[i] = 0.f; cx[i] = 0.f; }
  if (gt < 16) sync[gt] = 0u;
}

// ================= kernel 2: wq = q @ whq^T + b (WMMA GEMM) =================
// wq stored [q][b][h], row m = q*64 + b. M=4096, N=256, K=256.
__global__ void k_wq(const float* question, const __bf16* whq_bf,
                     const float* whq_b, float* wq) {
  int lane = threadIdx.x & 31, wave = threadIdx.x >> 5;
  int tile = blockIdx.x * (blockDim.x >> 5) + wave;   // 0..4095
  int mt = tile >> 4, nt = tile & 15;
  int l15 = lane & 15;
  int m = mt * 16 + l15;
  int b = m & 63, q = m >> 6;
  const float* arow = question + (size_t)b * (QQ * HH) + (size_t)q * HH;
  int khalf = (lane >> 4) * 8;
  int kh16  = (lane >> 4) * 16;
  int hi8   = (lane >> 4) * 8;
  int col = nt * 16 + l15;
  const __bf16* brow = whq_bf + (size_t)col * HH;
  v8f acc = {};
  for (int kt = 0; kt < 8; ++kt)
    acc = wmma_bf16(frag_a_f32(arow + kt * 32, khalf),
                    frag_b_bf(brow + kt * 32 + kh16), acc);
  float bias = whq_b[col];
#pragma unroll
  for (int r = 0; r < 8; ++r)
    wq[(size_t)(mt * 16 + r + hi8) * HH + col] = acc[r] + bias;
}

// ================= kernel 3: bidirectional match scan =================
// grid = 2*NB blocks; blockIdx.x/NB selects direction; NB blocks cooperate
// through a software grid barrier; each direction runs PP sequential steps.
__global__ void __launch_bounds__(NT, 1) k_scan(
    const float* passage, const float* question, const float* wq,
    const float* wvec, const float* wbp,
    const float* whp_b, const float* whr_b,
    const float* fw_bih, const float* fw_bhh,
    const float* bw_bih, const float* bw_bhh,
    const __bf16* whp_bf, const __bf16* whr_bf,
    const __bf16* Wih_all, const __bf16* Whh_all,
    float* hx_all, float* cx_all, float* base_all, float* attn_all,
    float* gates_all, float* scores_all, unsigned* sync, float* out) {

  const int d    = blockIdx.x / NB;        // 0 = forward, 1 = backward
  const int blk  = blockIdx.x % NB;
  const int tid  = threadIdx.x;
  const int lane = tid & 31, wave = tid >> 5;
  const int gw   = blk * NWAVE + wave;     // 0..GW-1 within direction
  const int gtid = blk * NT + tid;         // 0..NB*NT-1 within direction
  const int khalf = (lane >> 4) * 8;
  const int kh16  = (lane >> 4) * 16;
  const int hi8   = (lane >> 4) * 8;
  const int l15   = lane & 15;

  float* hx     = hx_all    + d * (BB * HH);
  float* cx     = cx_all    + d * (BB * HH);
  float* base   = base_all  + d * (BB * HH);
  float* attn   = attn_all  + d * (BB * HH);
  float* gates  = gates_all + d * (BB * 4 * HH);
  float* scores = scores_all + d * (BB * QQ);
  unsigned* syn = sync + d * 8;
  const __bf16* Wih = Wih_all + (size_t)d * (1024 * 512);
  const __bf16* Whh = Whh_all + (size_t)d * (1024 * 256);
  const float* bih = d ? bw_bih : fw_bih;
  const float* bhh = d ? bw_bhh : fw_bhh;
  const float wv_b = wbp[0];

  __shared__ float s_alpha[BB * QQ];   // 16 KB

  for (int step = 0; step < PP; ++step) {
    const int t = d ? (PP - 1 - step) : step;

    // -- Phase A: base = p_t @ whp^T + h @ whr^T + whp_b + whr_b  (64x256) --
    for (int tile = gw; tile < 64; tile += GW) {
      int mt = tile >> 4, nt = tile & 15;
      int m = mt * 16 + l15;               // batch row
      int col = nt * 16 + l15;
      const float* prow = passage + (size_t)m * (PP * HH) + (size_t)t * HH;
      const float* hrow = hx + (size_t)m * HH;
      const __bf16* bp = whp_bf + (size_t)col * HH;
      const __bf16* br = whr_bf + (size_t)col * HH;
      v8f acc = {};
      for (int kt = 0; kt < 8; ++kt)
        acc = wmma_bf16(frag_a_f32(prow + kt * 32, khalf),
                        frag_b_bf(bp + kt * 32 + kh16), acc);
      for (int kt = 0; kt < 8; ++kt)
        acc = wmma_bf16(frag_a_f32(hrow + kt * 32, khalf),
                        frag_b_bf(br + kt * 32 + kh16), acc);
      float bias = whp_b[col] + whr_b[col];
#pragma unroll
      for (int r = 0; r < 8; ++r)
        base[(size_t)(mt * 16 + r + hi8) * HH + col] = acc[r] + bias;
    }
    dir_barrier(syn, NB);

    // -- Phase B: scores[b][q] = sum_h w[h]*tanh(wq[q][b][h] + base[b][h]) --
    for (int pr = gw; pr < BB * QQ; pr += GW) {
      int b = pr >> 6, q = pr & 63;
      const float* wqrow = wq + ((size_t)q * BB + b) * HH;
      const float* brow  = base + (size_t)b * HH;
      float acc = 0.f;
#pragma unroll
      for (int j = 0; j < 8; ++j) {
        int h = lane + 32 * j;
        acc += wvec[h] * fast_tanh(wqrow[h] + brow[h]);
      }
#pragma unroll
      for (int off = 16; off; off >>= 1) acc += __shfl_xor(acc, off, 32);
      if (lane == 0) scores[pr] = acc + wv_b;
    }
    dir_barrier(syn, NB);

    // -- Phase C: softmax over Q (redundant per block, alpha into LDS) --
    for (int b = wave; b < BB; b += NWAVE) {
      float s0 = scores[b * 64 + lane];
      float s1 = scores[b * 64 + 32 + lane];
      float mx = fmaxf(s0, s1);
#pragma unroll
      for (int off = 16; off; off >>= 1) mx = fmaxf(mx, __shfl_xor(mx, off, 32));
      float e0 = fast_exp(s0 - mx), e1 = fast_exp(s1 - mx);
      float sm = e0 + e1;
#pragma unroll
      for (int off = 16; off; off >>= 1) sm += __shfl_xor(sm, off, 32);
      float inv = 1.f / sm;
      s_alpha[b * 64 + lane]      = e0 * inv;
      s_alpha[b * 64 + 32 + lane] = e1 * inv;
    }
    __syncthreads();

    // -- Phase D: attn[b][h] = sum_q alpha[b][q] * question[b][q][h] --
    for (int idx = gtid; idx < BB * HH; idx += NB * NT) {
      int b = idx >> 8, h = idx & 255;
      const float* qb = question + (size_t)b * (QQ * HH) + h;
      float acc = 0.f;
#pragma unroll 4
      for (int q = 0; q < QQ; ++q) acc += s_alpha[b * 64 + q] * qb[(size_t)q * HH];
      attn[idx] = acc;
    }
    dir_barrier(syn, NB);

    // -- Phase E: gates = [p_t | attn] @ Wih^T + h @ Whh^T   (64x1024) --
    for (int tile = gw; tile < 256; tile += GW) {
      int mt = tile >> 6, nt = tile & 63;
      int m = mt * 16 + l15;
      int col = nt * 16 + l15;
      const float* prow = passage + (size_t)m * (PP * HH) + (size_t)t * HH;
      const float* arow = attn + (size_t)m * HH;
      const float* hrow = hx + (size_t)m * HH;
      const __bf16* wih = Wih + (size_t)col * 512;
      const __bf16* whh = Whh + (size_t)col * 256;
      v8f acc = {};
      for (int kt = 0; kt < 8; ++kt)            // k in [0,256): p_t
        acc = wmma_bf16(frag_a_f32(prow + kt * 32, khalf),
                        frag_b_bf(wih + kt * 32 + kh16), acc);
      for (int kt = 0; kt < 8; ++kt)            // k in [256,512): attn
        acc = wmma_bf16(frag_a_f32(arow + kt * 32, khalf),
                        frag_b_bf(wih + 256 + kt * 32 + kh16), acc);
      for (int kt = 0; kt < 8; ++kt)            // recurrent term
        acc = wmma_bf16(frag_a_f32(hrow + kt * 32, khalf),
                        frag_b_bf(whh + kt * 32 + kh16), acc);
#pragma unroll
      for (int r = 0; r < 8; ++r)
        gates[(size_t)(mt * 16 + r + hi8) * 1024 + col] = acc[r];
    }
    dir_barrier(syn, NB);

    // -- Phase F: LSTM cell update + write output slice --
    for (int idx = gtid; idx < BB * HH; idx += NB * NT) {
      int b = idx >> 8, j = idx & 255;
      const float* g = gates + (size_t)b * 1024;
      float gi = g[j]       + bih[j]       + bhh[j];
      float gf = g[256 + j] + bih[256 + j] + bhh[256 + j];
      float gg = g[512 + j] + bih[512 + j] + bhh[512 + j];
      float go = g[768 + j] + bih[768 + j] + bhh[768 + j];
      float c = fast_sigmoid(gf) * cx[idx] + fast_sigmoid(gi) * fast_tanh(gg);
      float h = fast_sigmoid(go) * fast_tanh(c);
      cx[idx] = c;
      hx[idx] = h;
      out[((size_t)t * BB + b) * (2 * HH) + (size_t)d * HH + j] = h;
    }
    dir_barrier(syn, NB);   // h/c visible before next step's Phase A/E
  }
}

// ============================ host launcher ============================
extern "C" void kernel_launch(void* const* d_in, const int* in_sizes, int n_in,
                              void* d_out, int out_size, void* d_ws, size_t ws_size,
                              hipStream_t stream) {
  const float* passage  = (const float*)d_in[0];
  const float* question = (const float*)d_in[1];
  const float* whq_w = (const float*)d_in[2];
  const float* whq_b = (const float*)d_in[3];
  const float* whp_w = (const float*)d_in[4];
  const float* whp_b = (const float*)d_in[5];
  const float* whr_w = (const float*)d_in[6];
  const float* whr_b = (const float*)d_in[7];
  const float* w_w   = (const float*)d_in[8];   // (1,H) -> vector of H
  const float* w_b   = (const float*)d_in[9];   // scalar
  const float* fw_Wih = (const float*)d_in[10];
  const float* fw_Whh = (const float*)d_in[11];
  const float* fw_bih = (const float*)d_in[12];
  const float* fw_bhh = (const float*)d_in[13];
  const float* bw_Wih = (const float*)d_in[14];
  const float* bw_Whh = (const float*)d_in[15];
  const float* bw_bih = (const float*)d_in[16];
  const float* bw_bhh = (const float*)d_in[17];
  float* out = (float*)d_out;

  // ---- scratch carve-up (~8.2 MB total) ----
  char* w = (char*)d_ws;
  size_t o = 0;
  auto alloc = [&](size_t bytes) -> char* {
    char* p = w + o;
    o += (bytes + 255) & ~(size_t)255;
    return p;
  };
  float*  wq      = (float*) alloc((size_t)4096 * 256 * 4);     // [q][b][h]
  __bf16* whq_bf  = (__bf16*)alloc((size_t)65536 * 2);
  __bf16* whp_bf  = (__bf16*)alloc((size_t)65536 * 2);
  __bf16* whr_bf  = (__bf16*)alloc((size_t)65536 * 2);
  __bf16* Wih_bf  = (__bf16*)alloc((size_t)2 * 524288 * 2);     // fw|bw
  __bf16* Whh_bf  = (__bf16*)alloc((size_t)2 * 262144 * 2);
  float*  hx      = (float*) alloc((size_t)2 * 16384 * 4);
  float*  cx      = (float*) alloc((size_t)2 * 16384 * 4);
  float*  basep   = (float*) alloc((size_t)2 * 16384 * 4);
  float*  attnp   = (float*) alloc((size_t)2 * 16384 * 4);
  float*  gates   = (float*) alloc((size_t)2 * 65536 * 4);
  float*  scoresg = (float*) alloc((size_t)2 * 4096 * 4);
  unsigned* sync  = (unsigned*)alloc(64);
  (void)ws_size; (void)in_sizes; (void)n_in; (void)out_size;

  k_prep<<<256, 256, 0, stream>>>(whq_w, whp_w, whr_w, fw_Wih, fw_Whh, bw_Wih,
                                  bw_Whh, whq_bf, whp_bf, whr_bf, Wih_bf, Whh_bf,
                                  hx, cx, sync);
  k_wq<<<512, 256, 0, stream>>>(question, whq_bf, whq_b, wq);
  k_scan<<<2 * NB, NT, 0, stream>>>(passage, question, wq, w_w, w_b, whp_b,
                                    whr_b, fw_bih, fw_bhh, bw_bih, bw_bhh,
                                    whp_bf, whr_bf, Wih_bf, Whh_bf, hx, cx,
                                    basep, attnp, gates, scoresg, sync, out);
}
